// AttentionZPooler_62122406969654
// MI455X (gfx1250) — matrix operand order repaired and checked by reading, not compiled, not run on hardware
//
#include <hip/hip_runtime.h>
#include <hip/hip_bf16.h>
#include <math.h>
#include <stdint.h>

typedef __attribute__((ext_vector_type(16))) _Float16 v16h;
typedef __attribute__((ext_vector_type(8)))  float    v8f;

#define NTOK 8192
#define MTOK 1024

// ---------------------------------------------------------------------------
// Pooling: X_tilde[m,d] = (sum_n G[n,m]*X[n,d]) / (sum_n G[n,m] + 1e-8)
// Coalesced across m; G (32MB) is the dominant HBM stream of the pipeline.
// ---------------------------------------------------------------------------
__global__ void pool_kernel(const float* __restrict__ G, const float* __restrict__ X,
                            float* __restrict__ Xt, int n, int m) {
  const int col = blockIdx.x * blockDim.x + threadIdx.x; // m index
  if (col >= m) return;
  float s = 0.f;
  float t[16];
#pragma unroll
  for (int d = 0; d < 16; ++d) t[d] = 0.f;
  for (int i = 0; i < n; ++i) {
    const float g = G[(size_t)i * m + col];
    s += g;
    const float* xr = X + (size_t)i * 16;
#pragma unroll
    for (int d = 0; d < 16; ++d) t[d] += g * xr[d];
  }
  const float inv = 1.f / (s + 1e-8f);
#pragma unroll
  for (int d = 0; d < 16; ++d) Xt[(size_t)col * 16 + d] = t[d] * inv;
}

// ---------------------------------------------------------------------------
// Residual MLP  [D, Hd, Hd, Hd, D] with ReLU; optional repeat(x,4) input
// (jnp.repeat: x_in[k] = in[k/4]).  Tiny FLOPs -> thread-per-row VALU kernel.
// ---------------------------------------------------------------------------
template <int D, int Hd, int REP>
__global__ void res_mlp_kernel(const float* __restrict__ in, int inLd,
                               float* __restrict__ out, int outLd,
                               const float* __restrict__ W1, const float* __restrict__ b1,
                               const float* __restrict__ W2, const float* __restrict__ b2,
                               const float* __restrict__ W3, const float* __restrict__ b3,
                               const float* __restrict__ W4, const float* __restrict__ b4,
                               int rows) {
  const int row = blockIdx.x * blockDim.x + threadIdx.x;
  if (row >= rows) return;
  const float* xr = in + (size_t)row * inLd;
  float h1[Hd], h2[Hd];
#pragma unroll
  for (int j = 0; j < Hd; ++j) {
    float acc = b1[j];
    for (int k = 0; k < D; ++k) acc += W1[j * D + k] * xr[REP ? (k >> 2) : k];
    h1[j] = fmaxf(acc, 0.f);
  }
#pragma unroll
  for (int j = 0; j < Hd; ++j) {
    float acc = b2[j];
#pragma unroll
    for (int k = 0; k < Hd; ++k) acc += W2[j * Hd + k] * h1[k];
    h2[j] = fmaxf(acc, 0.f);
  }
#pragma unroll
  for (int j = 0; j < Hd; ++j) {
    float acc = b3[j];
#pragma unroll
    for (int k = 0; k < Hd; ++k) acc += W3[j * Hd + k] * h2[k];
    h1[j] = fmaxf(acc, 0.f);
  }
  for (int j = 0; j < D; ++j) {
    float acc = b4[j];
#pragma unroll
    for (int k = 0; k < Hd; ++k) acc += W4[j * Hd + k] * h1[k];
    out[(size_t)row * outLd + j] = xr[REP ? (j >> 2) : j] + acc;
  }
}

// ---------------------------------------------------------------------------
// Generic WMMA GEMM:  O = A(rows x K, f32, lda) @ W^T (W: out x K, f32) + bias
// One wave per 16x16 output tile, K in steps of 32 via v_wmma_f32_16x16x32_f16.
// Outputs (any non-null): f16 row-major, f16 transposed (for V^T), f32 row-major.
// ---------------------------------------------------------------------------
__global__ __launch_bounds__(32)
void gemm16_kernel(const float* __restrict__ A, int lda,
                   const float* __restrict__ W, const float* __restrict__ bias, int K,
                   int rowsTotal,
                   _Float16* __restrict__ outH, int ldoH,
                   _Float16* __restrict__ outT,
                   float* __restrict__ outF, int ldF) {
  const int lane = threadIdx.x;
  const int qr = lane & 15;       // tile row (A) / tile col (B,C)
  const int hi = lane >> 4;       // lane half
  const int kb = hi * 8;          // A-fragment K base within half
  const int r0 = blockIdx.x * 16;
  const int c0 = blockIdx.y * 16;

  v8f c;
#pragma unroll
  for (int r = 0; r < 8; ++r) c[r] = 0.f;

  const float* arow = A + (size_t)(r0 + qr) * lda;
  const float* wrow = W + (size_t)(c0 + qr) * K;

  for (int k0 = 0; k0 < K; k0 += 32) {
    v16h a, b;
#pragma unroll
    for (int e = 0; e < 16; ++e) {
      const int k = k0 + ((e < 8) ? (kb + e) : (16 + kb + (e - 8)));
      a[e] = (_Float16)arow[k];
    }
#pragma unroll
    for (int e = 0; e < 16; ++e) b[e] = (_Float16)wrow[k0 + ((e < 8) ? (kb + e) : (16 + kb + (e - 8)))];
    c = __builtin_amdgcn_wmma_f32_16x16x32_f16(false, a, false, b, (short)0, c, false, false);
  }

  const float bb = bias[c0 + qr];
#pragma unroll
  for (int r = 0; r < 8; ++r) {
    const int row = r0 + hi * 8 + r;
    const int col = c0 + qr;
    const float v = c[r] + bb;
    if (outH) outH[(size_t)row * ldoH + col] = (_Float16)v;
    if (outT) outT[(size_t)col * rowsTotal + row] = (_Float16)v;
    if (outF) outF[(size_t)row * ldF + col] = v;
  }
}

// ---------------------------------------------------------------------------
// Flash attention, 4 waves / 128 threads per block, 64 query rows per block,
// one head per blockIdx.y.  Key tile = 32, double-buffered in LDS:
//   - next K (32 x HD) and V^T (HD x 32) tiles are fetched with
//     global_load_async_to_lds_b128 (ASYNCcnt path, no VGPR round-trip)
//     while the current tile feeds the WMMAs;
//   - one s_wait_asynccnt 0 + one block barrier per iteration.
// Online softmax in registers; P staged wave-locally through LDS to convert
// C-layout scores -> A-layout operand for the P@V WMMA.  The nq x nk score
// matrix (512MB for SAB) never exists in HBM.
// ---------------------------------------------------------------------------
template <int HD>
__global__ __launch_bounds__(128)
void flash_kernel(const _Float16* __restrict__ Q, const _Float16* __restrict__ K,
                  const _Float16* __restrict__ Vt, float* __restrict__ O,
                  int nk, int qs, float scale) {
  constexpr int NF = (HD + 31) / 32;   // A fragments along head dim
  constexpr int NG = HD / 16;          // output column groups
  constexpr int KCH = 32 * HD / 16;    // 16B chunks in K tile
  constexpr int VCH = HD * 32 / 16;    // 16B chunks in V tile

  __shared__ alignas(16) _Float16 Kt[2][32 * HD];    // [buf][key][k]
  __shared__ alignas(16) _Float16 Vs[2][HD * 32];    // [buf][d_local][key]
  __shared__ alignas(16) _Float16 P[4 * 16 * 32];    // per-wave 16x32 P tile

  const int tid = threadIdx.x;
  const int lane = tid & 31;
  const int wave = tid >> 5;
  const int qr = lane & 15;
  const int hi = lane >> 4;
  const int kb = hi * 8;
  const int q0 = blockIdx.x * 64 + wave * 16;
  const int head = blockIdx.y;
  _Float16* Pw = P + wave * (16 * 32);

  const _Float16* Kbase = K + (size_t)head * HD;
  const _Float16* Vbase = Vt + (size_t)head * HD * nk;

  // async stage of one 32-key K/V tile into LDS buffer `buf`
  auto stage = [&](int kt, int buf) {
    for (int c = tid; c < KCH; c += 128) {
      const int key = c / (HD / 16);
      const int ko = (c % (HD / 16)) * 16;
      const _Float16* src = Kbase + (size_t)(kt + key) * qs + ko;
      const uint32_t dst = (uint32_t)(uintptr_t)(&Kt[buf][key * HD + ko]);
      asm volatile("global_load_async_to_lds_b128 %0, %1, off"
                   :: "v"(dst), "v"(src) : "memory");
    }
    for (int c = tid; c < VCH; c += 128) {
      const int dl = c >> 1;
      const int co = (c & 1) * 16;
      const _Float16* src = Vbase + (size_t)dl * nk + kt + co;
      const uint32_t dst = (uint32_t)(uintptr_t)(&Vs[buf][dl * 32 + co]);
      asm volatile("global_load_async_to_lds_b128 %0, %1, off"
                   :: "v"(dst), "v"(src) : "memory");
    }
  };

  // Q tile in WMMA A layout (zero-pad K>=16 when HD==16)
  v16h aq[NF];
  const _Float16* Qp = Q + (size_t)(q0 + qr) * qs + head * HD;
#pragma unroll
  for (int f = 0; f < NF; ++f) {
#pragma unroll
    for (int e = 0; e < 16; ++e) {
      const int k = f * 32 + ((e < 8) ? (kb + e) : (16 + kb + (e - 8)));
      float v = 0.f;
      if (!(HD == 16 && k >= 16)) v = (float)Qp[k];
      aq[f][e] = (_Float16)v;
    }
  }

  float m[8], l[8];
  v8f o[NG];
#pragma unroll
  for (int r = 0; r < 8; ++r) { m[r] = -3.0e38f; l[r] = 0.f; }
#pragma unroll
  for (int g = 0; g < NG; ++g)
#pragma unroll
    for (int r = 0; r < 8; ++r) o[g][r] = 0.f;

  stage(0, 0);  // prologue: tile 0 in flight

  for (int kt = 0; kt < nk; kt += 32) {
    const int buf = (kt >> 5) & 1;
    // Drain this wave's async loads; barrier then guarantees every wave's
    // loads for `buf` are complete AND every wave finished reading buf^1
    // last iteration, so re-issuing into buf^1 below is race-free.
    asm volatile("s_wait_asynccnt 0" ::: "memory");
    __syncthreads();
    if (kt + 32 < nk) stage(kt + 32, buf ^ 1);

    // ---- scores: two 16x16 tiles (keys kt..+15, kt+16..+31), B from LDS
    v8f c2[2];
#pragma unroll
    for (int s = 0; s < 2; ++s) {
#pragma unroll
      for (int r = 0; r < 8; ++r) c2[s][r] = 0.f;
      const _Float16* Kp = &Kt[buf][(s * 16 + qr) * HD];
#pragma unroll
      for (int f = 0; f < NF; ++f) {
        v16h bk;
#pragma unroll
        for (int e = 0; e < 16; ++e) {
          const int k = f * 32 + ((e < 8) ? (kb + e) : (16 + kb + (e - 8)));
          float v = 0.f;
          if (!(HD == 16 && k >= 16)) v = (float)Kp[k];
          bk[e] = (_Float16)v;
        }
        c2[s] = __builtin_amdgcn_wmma_f32_16x16x32_f16(false, aq[f], false, bk, (short)0, c2[s], false, false);
      }
    }

    // ---- online softmax (row stats replicated across each 16-lane half)
#pragma unroll
    for (int r = 0; r < 8; ++r) {
      const float s0 = c2[0][r] * scale;
      const float s1 = c2[1][r] * scale;
      float t = fmaxf(s0, s1);
#pragma unroll
      for (int off = 1; off < 16; off <<= 1) t = fmaxf(t, __shfl_xor(t, off, 32));
      const float mn = fmaxf(m[r], t);
      const float ef = __expf(m[r] - mn);
      const float p0 = __expf(s0 - mn);
      const float p1 = __expf(s1 - mn);
      float ps = p0 + p1;
#pragma unroll
      for (int off = 1; off < 16; off <<= 1) ps += __shfl_xor(ps, off, 32);
      l[r] = l[r] * ef + ps;
      m[r] = mn;
#pragma unroll
      for (int g = 0; g < NG; ++g) o[g][r] *= ef;
      const int prow = hi * 8 + r;
      Pw[prow * 32 + qr] = (_Float16)p0;
      Pw[prow * 32 + 16 + qr] = (_Float16)p1;
    }
    // wave-local P store->load: DS ops from one wave complete in order,
    // so no block barrier is needed here.

    // ---- P (16x32) back as A-fragment; V tile as B-fragment (both from LDS)
    v16h ap;
#pragma unroll
    for (int e = 0; e < 16; ++e) {
      const int k = hi * 16 + e;
      ap[e] = Pw[qr * 32 + k];
    }
#pragma unroll
    for (int g = 0; g < NG; ++g) {
      v16h bv;
      const _Float16* Vp = &Vs[buf][(g * 16 + qr) * 32 + hi * 16];
#pragma unroll
      for (int e = 0; e < 16; ++e) bv[e] = Vp[e];
      o[g] = __builtin_amdgcn_wmma_f32_16x16x32_f16(false, ap, false, bv, (short)0, o[g], false, false);
    }
  }

#pragma unroll
  for (int r = 0; r < 8; ++r) {
    const float inv = 1.f / l[r];
    const int prow = hi * 8 + r;
#pragma unroll
    for (int g = 0; g < NG; ++g)
      O[(size_t)(q0 + prow) * qs + head * HD + g * 16 + qr] = o[g][r] * inv;
  }
}

// ---------------------------------------------------------------------------
extern "C" void kernel_launch(void* const* d_in, const int* in_sizes, int n_in,
                              void* d_out, int out_size, void* d_ws, size_t ws_size,
                              hipStream_t stream) {
  (void)in_sizes; (void)n_in; (void)out_size; (void)ws_size;
  const int N = NTOK, M = MTOK;

  const float* X      = (const float*)d_in[0];   // N x 16
  const float* H      = (const float*)d_in[1];   // M x 32
  const float* G      = (const float*)d_in[2];   // N x M
  const float* fcxW1  = (const float*)d_in[3];   const float* fcxB1 = (const float*)d_in[4];
  const float* fcxW2  = (const float*)d_in[5];   const float* fcxB2 = (const float*)d_in[6];
  const float* fcxW3  = (const float*)d_in[7];   const float* fcxB3 = (const float*)d_in[8];
  const float* fcxW4  = (const float*)d_in[9];   const float* fcxB4 = (const float*)d_in[10];
  const float* fc1W1  = (const float*)d_in[11];  const float* fc1B1 = (const float*)d_in[12];
  const float* fc1W2  = (const float*)d_in[13];  const float* fc1B2 = (const float*)d_in[14];
  const float* fc1W3  = (const float*)d_in[15];  const float* fc1B3 = (const float*)d_in[16];
  const float* fc1W4  = (const float*)d_in[17];  const float* fc1B4 = (const float*)d_in[18];
  const float* fc2W1  = (const float*)d_in[19];  const float* fc2B1 = (const float*)d_in[20];
  const float* fc2W2  = (const float*)d_in[21];  const float* fc2B2 = (const float*)d_in[22];
  const float* fc2W3  = (const float*)d_in[23];  const float* fc2B3 = (const float*)d_in[24];
  const float* fc2W4  = (const float*)d_in[25];  const float* fc2B4 = (const float*)d_in[26];
  const float* linW   = (const float*)d_in[27];  const float* linB  = (const float*)d_in[28];
  const float* mabQW  = (const float*)d_in[29];
  const float* mabKW  = (const float*)d_in[30];
  const float* mabVW  = (const float*)d_in[31];
  const float* mabB   = (const float*)d_in[32];
  const float* mabOW  = (const float*)d_in[33];
  const float* mabOB  = (const float*)d_in[34];
  const float* sabW   = (const float*)d_in[35];
  const float* sabB   = (const float*)d_in[36];
  const float* sabOW  = (const float*)d_in[37];
  const float* sabOB  = (const float*)d_in[38];
  float* out = (float*)d_out;

  // workspace carve-up (aliased: Zc doubles as Z2, Z1 doubles as Z3)
  char* base = (char*)d_ws; size_t off = 0;
  auto alloc = [&](size_t bytes) { char* p = base + off; off += (bytes + 255) & ~(size_t)255; return p; };
  float*     Xt   = (float*)alloc((size_t)M * 16 * 4);
  float*     Zc   = (float*)alloc((size_t)N * 128 * 4);  // [X_emb | mab_out] -> later Z2
  float*     Z1   = (float*)alloc((size_t)N * 128 * 4);  // fc1 out -> later Z3
  float*     XtE  = (float*)alloc((size_t)M * 64 * 4);
  float*     Om   = (float*)alloc((size_t)N * 64 * 4);
  float*     Os   = (float*)alloc((size_t)N * 128 * 4);
  _Float16*  Qm   = (_Float16*)alloc((size_t)N * 64 * 2);
  _Float16*  Km   = (_Float16*)alloc((size_t)M * 64 * 2);
  _Float16*  VmT  = (_Float16*)alloc((size_t)M * 64 * 2);
  _Float16*  Qs   = (_Float16*)alloc((size_t)N * 128 * 2);
  _Float16*  Ks   = (_Float16*)alloc((size_t)N * 128 * 2);
  _Float16*  VsT  = (_Float16*)alloc((size_t)N * 128 * 2);

  // 1) pooling  (reads G once, coalesced)
  pool_kernel<<<dim3(M / 64), dim3(64), 0, stream>>>(G, X, Xt, N, M);

  // 2) embeddings (repeat x4 + fcx res-MLP); X_emb lands in Zc[:, 0:64]
  res_mlp_kernel<64, 8, 1><<<dim3(N / 256), dim3(256), 0, stream>>>(
      X, 16, Zc, 128, fcxW1, fcxB1, fcxW2, fcxB2, fcxW3, fcxB3, fcxW4, fcxB4, N);
  res_mlp_kernel<64, 8, 1><<<dim3(M / 256), dim3(256), 0, stream>>>(
      Xt, 16, XtE, 64, fcxW1, fcxB1, fcxW2, fcxB2, fcxW3, fcxB3, fcxW4, fcxB4, M);

  // 3) MAB projections (WMMA; V written transposed)
  gemm16_kernel<<<dim3(N / 16, 4), dim3(32), 0, stream>>>(
      Zc, 128, mabQW, mabB + 0, 64, N, Qm, 64, nullptr, nullptr, 0);
  gemm16_kernel<<<dim3(M / 16, 4), dim3(32), 0, stream>>>(
      XtE, 64, mabKW, mabB + 64, 64, M, Km, 64, nullptr, nullptr, 0);
  gemm16_kernel<<<dim3(M / 16, 4), dim3(32), 0, stream>>>(
      H, 32, mabVW, mabB + 128, 32, M, nullptr, 0, VmT, nullptr, 0);

  // 4) MAB flash attention (4 heads, hd=16, keys = M); 64 q-rows per block
  flash_kernel<16><<<dim3(N / 64, 4), dim3(128), 0, stream>>>(Qm, Km, VmT, Om, M, 64, 0.25f);

  // 5) MAB output projection -> Zc[:, 64:128]
  gemm16_kernel<<<dim3(N / 16, 4), dim3(32), 0, stream>>>(
      Om, 64, mabOW, mabOB, 64, N, nullptr, 0, nullptr, Zc + 64, 128);

  // 6) fc1 res-MLP
  res_mlp_kernel<128, 16, 0><<<dim3(N / 256), dim3(256), 0, stream>>>(
      Zc, 128, Z1, 128, fc1W1, fc1B1, fc1W2, fc1B2, fc1W3, fc1B3, fc1W4, fc1B4, N);

  // 7) SAB projections from packed sab_w (q|k|v rows of 128)
  gemm16_kernel<<<dim3(N / 16, 8), dim3(32), 0, stream>>>(
      Z1, 128, sabW, sabB, 128, N, Qs, 128, nullptr, nullptr, 0);
  gemm16_kernel<<<dim3(N / 16, 8), dim3(32), 0, stream>>>(
      Z1, 128, sabW + 128 * 128, sabB + 128, 128, N, Ks, 128, nullptr, nullptr, 0);
  gemm16_kernel<<<dim3(N / 16, 8), dim3(32), 0, stream>>>(
      Z1, 128, sabW + 2 * 128 * 128, sabB + 256, 128, N, nullptr, 0, VsT, nullptr, 0);

  // 8) SAB flash attention (2 heads, hd=64, keys = N) — the 34 GFLOP hot spot
  flash_kernel<64><<<dim3(N / 64, 2), dim3(128), 0, stream>>>(Qs, Ks, VsT, Os, N, 128, 0.125f);

  // 9) SAB output projection -> Z2 (aliases Zc)
  float* Z2 = Zc;
  gemm16_kernel<<<dim3(N / 16, 8), dim3(32), 0, stream>>>(
      Os, 128, sabOW, sabOB, 128, N, nullptr, 0, nullptr, Z2, 128);

  // 10) fc2 res-MLP -> Z3 (aliases Z1)
  float* Z3 = Z1;
  res_mlp_kernel<128, 16, 0><<<dim3(N / 256), dim3(256), 0, stream>>>(
      Z2, 128, Z3, 128, fc2W1, fc2B1, fc2W2, fc2B2, fc2W3, fc2B3, fc2W4, fc2B4, N);

  // 11) final linear 128 -> 32 into d_out
  gemm16_kernel<<<dim3(N / 16, 2), dim3(32), 0, stream>>>(
      Z3, 128, linW, linB, 128, N, nullptr, 0, nullptr, out, 32);
}
